// TransformerTransferGSCPredictor_76476187673013
// MI455X (gfx1250) — matrix operand order repaired and checked by reading, not compiled, not run on hardware
//
#include <hip/hip_runtime.h>
#include <hip/hip_bf16.h>
#include <cstddef>
#include <cstdint>

// ---------------- model constants ----------------
#define HIDDEN 1024
#define NEI    4096      // N_ENT_IMG
#define NPI    16384     // N_PRED_IMG
#define NOC    151       // N_OBJ_CLS
#define NRC    51        // N_REL_CLS
#define EMB    300
#define H2     512       // HIDDEN/2
#define H4     256       // HIDDEN/4

// ---------------- vector types ----------------
typedef __attribute__((ext_vector_type(16))) __bf16 v16bf;
typedef __attribute__((ext_vector_type(8)))  float  v8f;
typedef unsigned int  u32x4 __attribute__((ext_vector_type(4)));
typedef unsigned int  u32x2 __attribute__((ext_vector_type(2)));
typedef float         f32x4 __attribute__((ext_vector_type(4)));

union V16U { v16bf v; u32x4 q[2]; };

__device__ __forceinline__ unsigned short f32_bf16(float f) {
  unsigned int u = __float_as_uint(f);
  unsigned int r = 0x7FFFu + ((u >> 16) & 1u);   // round-to-nearest-even
  return (unsigned short)((u + r) >> 16);
}

// ---------------- WMMA GEMM ----------------
// C[M,N](f32) = act( A x B + bias (+C if ACC) )
//   AT=false : A is f32 [M,K] row-major
//   AT=true  : A is f32 [K,M] row-major (computes A^T x B)
//   BN32=false: B is bf16 [N,K] row-major (pre-transposed weight)
//   BN32=true : B is f32  [K,N] row-major (activation, transposed while staging)
// Tile: 128x128x64, 8 waves, wave tile 32x64 -> 2 sub-steps of 2x4
// v_wmma_f32_16x16x32_bf16 per staged tile (16 WMMA / stage / wave).
// Full bf16 B tiles are staged with GLOBAL_LOAD_ASYNC_TO_LDS_B128 (ASYNCcnt).
#define TBM 128
#define TBN 128
#define TBK 64
#define LDA_S 72   // padded row stride (ushorts): 144B rows -> 16B-aligned,
#define LDB_S 72   // conflict-free 16B operand reads (gcd(36,64)=4 -> 16 banks)

template<bool AT, bool BN32, int ACT, bool ACC>
__global__ __launch_bounds__(256)
void k_gemm(const float* __restrict__ A, const void* __restrict__ Bv,
            const float* __restrict__ bias, float* __restrict__ C,
            int M, int N, int K, int ldc)
{
  __shared__ unsigned short As[TBM * LDA_S];
  __shared__ unsigned short Bs[TBN * LDB_S];

  const int tid  = threadIdx.x;
  const int lane = tid & 31;
  const int wv   = tid >> 5;
  const int wm   = wv >> 1;            // 0..3 : 32-row slice
  const int wn   = wv & 1;             // 0..1 : 64-col slice
  const int m0   = blockIdx.y * TBM;
  const int n0   = blockIdx.x * TBN;

  const bool a_vec = (((AT ? M : K) & 3) == 0);
  const bool b_vec = BN32 ? ((N & 3) == 0) : ((K & 7) == 0);

  // WMMA operand addressing (per ISA 7.12.2, wave32)
  const int ml  = lane & 15;
  const int khA = (lane >> 4) << 3;    // 0 or 8  (ushorts)
  const int khB = (lane >> 4) << 3;    // 0 or 8  (ushorts)

  v8f acc[2][4];
#pragma unroll
  for (int i = 0; i < 2; ++i)
#pragma unroll
    for (int j = 0; j < 4; ++j)
#pragma unroll
      for (int e = 0; e < 8; ++e) acc[i][j][e] = 0.0f;

  for (int k0 = 0; k0 < K; k0 += TBK) {
    // ---- stage A -> As[m][k] (bf16, f32->bf16 converted in flight) ----
    if (!AT) {
#pragma unroll
      for (int it = 0; it < 8; ++it) {
        int f = tid + it * 256;          // 2048 float4 slots (128 x 64)
        int row = f >> 4, c4 = (f & 15) << 2;
        int gm = m0 + row, gk = k0 + c4;
        f32x4 v = {0.f, 0.f, 0.f, 0.f};
        if (gm < M) {
          if (a_vec && (gk + 4 <= K)) v = *(const f32x4*)(A + (size_t)gm * K + gk);
          else
            for (int j = 0; j < 4; ++j)
              if (gk + j < K) v[j] = A[(size_t)gm * K + gk + j];
        }
        u32x2 pk;
        pk[0] = (unsigned)f32_bf16(v[0]) | ((unsigned)f32_bf16(v[1]) << 16);
        pk[1] = (unsigned)f32_bf16(v[2]) | ((unsigned)f32_bf16(v[3]) << 16);
        *(u32x2*)&As[row * LDA_S + c4] = pk;
      }
    } else {
#pragma unroll
      for (int it = 0; it < 8; ++it) {
        int f = tid + it * 256;          // [64 k][128 m]
        int kr = f >> 5, mc = (f & 31) << 2;
        int gk = k0 + kr, gm = m0 + mc;
        f32x4 v = {0.f, 0.f, 0.f, 0.f};
        if (gk < K) {
          if (a_vec && (gm + 4 <= M)) v = *(const f32x4*)(A + (size_t)gk * M + gm);
          else
            for (int j = 0; j < 4; ++j)
              if (gm + j < M) v[j] = A[(size_t)gk * M + gm + j];
        }
#pragma unroll
        for (int j = 0; j < 4; ++j) As[(mc + j) * LDA_S + kr] = f32_bf16(v[j]);
      }
    }
    // ---- stage B -> Bs[n][k] (bf16) ----
    bool asyncB = false;
    if (!BN32) {
      const unsigned short* Bp = (const unsigned short*)Bv; // bf16 [N,K]
      // Workgroup-uniform fast path: full tile, 16B-aligned rows -> async DMA to LDS.
      asyncB = ((n0 + TBN) <= N) && ((K & 7) == 0) && ((k0 + TBK) <= K);
      if (asyncB) {
#pragma unroll
        for (int it = 0; it < 4; ++it) {
          int f = tid + it * 256;         // 1024 16B groups (128 x 64 bf16)
          int row = f >> 3, c8 = (f & 7) << 3;
          int gn = n0 + row, gk = k0 + c8;
          const void* gp = (const void*)(Bp + (size_t)gn * K + gk);
          unsigned ldsa = (unsigned)(uintptr_t)&Bs[row * LDB_S + c8];
          asm volatile("global_load_async_to_lds_b128 %0, %1, off"
                       :: "v"(ldsa), "v"(gp) : "memory");
        }
      } else {
#pragma unroll
        for (int it = 0; it < 4; ++it) {
          int f = tid + it * 256;
          int row = f >> 3, c8 = (f & 7) << 3;
          int gn = n0 + row, gk = k0 + c8;
          u32x4 tv = {0u, 0u, 0u, 0u};
          unsigned short* t8 = (unsigned short*)&tv;
          if (gn < N) {
            if (b_vec && (gk + 8 <= K)) tv = *(const u32x4*)(Bp + (size_t)gn * K + gk);
            else
              for (int j = 0; j < 8; ++j)
                if (gk + j < K) t8[j] = Bp[(size_t)gn * K + gk + j];
          }
          *(u32x4*)&Bs[row * LDB_S + c8] = tv;
        }
      }
    } else {
      const float* Bp = (const float*)Bv;  // f32 [K,N]
#pragma unroll
      for (int it = 0; it < 8; ++it) {
        int f = tid + it * 256;            // [64 k][128 n]
        int kr = f >> 5, nc = (f & 31) << 2;
        int gk = k0 + kr, gn = n0 + nc;
        f32x4 v = {0.f, 0.f, 0.f, 0.f};
        if (gk < K) {
          if (b_vec && (gn + 4 <= N)) v = *(const f32x4*)(Bp + (size_t)gk * N + gn);
          else
            for (int j = 0; j < 4; ++j)
              if (gn + j < N) v[j] = Bp[(size_t)gk * N + gn + j];
        }
#pragma unroll
        for (int j = 0; j < 4; ++j) Bs[(nc + j) * LDB_S + kr] = f32_bf16(v[j]);
      }
    }
    if (asyncB) asm volatile("s_wait_asynccnt 0x0" ::: "memory");
    __syncthreads();

    // ---- WMMA: two 32-K sub-steps, 2x4 tiles per wave each ----
#pragma unroll
    for (int ks = 0; ks < 2; ++ks) {
      V16U a[2], b[4];
#pragma unroll
      for (int mi = 0; mi < 2; ++mi) {
        int r = (wm * 32 + mi * 16 + ml) * LDA_S + ks * 32;
        a[mi].q[0] = *(const u32x4*)&As[r + khA];        // K 0..7   (+khalf*8)
        a[mi].q[1] = *(const u32x4*)&As[r + 16 + khA];   // K 16..23 (+khalf*8)
      }
#pragma unroll
      for (int ni = 0; ni < 4; ++ni) {
        int r = (wn * 64 + ni * 16 + ml) * LDB_S + ks * 32 + khB;
        b[ni].q[0] = *(const u32x4*)&Bs[r];
        b[ni].q[1] = *(const u32x4*)&Bs[r + 16];
      }
#pragma unroll
      for (int mi = 0; mi < 2; ++mi)
#pragma unroll
        for (int ni = 0; ni < 4; ++ni)
          acc[mi][ni] = __builtin_amdgcn_wmma_f32_16x16x32_bf16(
              false, a[mi].v, false, b[ni].v, (short)0, acc[mi][ni], false, false);
    }
    __syncthreads();
  }

  // ---- epilogue ----
  const int moff = (lane >> 4) << 3;   // 0 or 8
#pragma unroll
  for (int mi = 0; mi < 2; ++mi)
#pragma unroll
    for (int ni = 0; ni < 4; ++ni) {
      int gn = n0 + wn * 64 + ni * 16 + ml;
      if (gn >= N) continue;
      float bv = bias ? bias[gn] : 0.0f;
#pragma unroll
      for (int r = 0; r < 8; ++r) {
        int gm = m0 + wm * 32 + mi * 16 + moff + r;
        if (gm >= M) continue;
        float v = acc[mi][ni][r] + bv;
        if (ACC) v += C[(size_t)gm * ldc + gn];
        if (ACT == 1) v = v > 0.0f ? v : 0.0f;
        C[(size_t)gm * ldc + gn] = v;
      }
    }
}

// ---------------- support kernels ----------------
__global__ void k_transpose_bf16(const float* __restrict__ src,
                                 unsigned short* __restrict__ dst, int K, int N) {
  // dst[n*K + k] = bf16(src[k*N + n])
  __shared__ float tile[32][33];
  int k0 = blockIdx.x * 32, n0 = blockIdx.y * 32;
  for (int i = threadIdx.y; i < 32; i += 8) {
    int k = k0 + i, n = n0 + threadIdx.x;
    tile[i][threadIdx.x] = (k < K && n < N) ? src[(size_t)k * N + n] : 0.f;
  }
  __syncthreads();
  for (int i = threadIdx.y; i < 32; i += 8) {
    int n = n0 + i, k = k0 + threadIdx.x;
    if (n < N && k < K) dst[(size_t)n * K + k] = f32_bf16(tile[threadIdx.x][i]);
  }
}

__global__ void k_convert_bf16(const float* __restrict__ s, unsigned short* __restrict__ d, int n) {
  int i = blockIdx.x * blockDim.x + threadIdx.x;
  if (i < n) d[i] = f32_bf16(s[i]);
}

__global__ void k_fill(float* __restrict__ p, float v, int n) {
  int i = blockIdx.x * blockDim.x + threadIdx.x;
  if (i < n) p[i] = v;
}

__global__ void k_softmax_rows(const float* __restrict__ src, float* __restrict__ dst,
                               int rows, int cols) {
  int row = blockIdx.x;
  if (row >= rows) return;
  int lane = threadIdx.x;                 // wave32: one wave per row
  const float* s = src + (size_t)row * cols;
  float mx = -3.4e38f;
  for (int c = lane; c < cols; c += 32) mx = fmaxf(mx, s[c]);
  for (int o = 16; o > 0; o >>= 1) mx = fmaxf(mx, __shfl_xor(mx, o, 32));
  float sum = 0.f;
  for (int c = lane; c < cols; c += 32) sum += __expf(s[c] - mx);
  for (int o = 16; o > 0; o >>= 1) sum += __shfl_xor(sum, o, 32);
  float inv = 1.f / sum;
  float* d = dst + (size_t)row * cols;
  for (int c = lane; c < cols; c += 32) d[c] = __expf(s[c] - mx) * inv;
}

__global__ void k_scatter_pairs(const float* __restrict__ src, const int* __restrict__ rp,
                                float* __restrict__ dst, int np, int cols, int ldd) {
  int i = blockIdx.x * blockDim.x + threadIdx.x;
  if (i >= np * cols) return;
  int p = i / cols, c = i - p * cols;
  float v = src[(size_t)p * cols + c];
  int s = rp[2 * p], o = rp[2 * p + 1];
  atomicAdd(&dst[(size_t)s * ldd + cols + c], v);       // b2 (subj)
  atomicAdd(&dst[(size_t)o * ldd + 2 * cols + c], v);   // b3 (obj)
}

__global__ void k_gather_pairs(const float* __restrict__ sie, const int* __restrict__ rp,
                               float* __restrict__ dst, int np, int cols, int ldd) {
  int i = blockIdx.x * blockDim.x + threadIdx.x;
  if (i >= np * cols) return;
  int p = i / cols, c = i - p * cols;
  int s = rp[2 * p], o = rp[2 * p + 1];
  dst[(size_t)p * ldd + cols + c]     = sie[(size_t)s * cols + c];
  dst[(size_t)p * ldd + 2 * cols + c] = sie[(size_t)o * cols + c];
}

__global__ void k_sig_mul(float* __restrict__ r, const float* __restrict__ node, int n) {
  int i = blockIdx.x * blockDim.x + threadIdx.x;
  if (i < n) {
    float s = 1.f / (1.f + __expf(-r[i]));
    r[i] = s * node[i];
  }
}

__global__ void k_gru_final(const float* __restrict__ Z, const float* __restrict__ H,
                            float* __restrict__ node, int n) {
  int i = blockIdx.x * blockDim.x + threadIdx.x;
  if (i < n) {
    float z = 1.f / (1.f + __expf(-Z[i]));
    node[i] = (1.f - z) * node[i] + z * tanhf(H[i]);
  }
}

// ---------------- host-side helpers ----------------
static void launch_gemm(hipStream_t st, bool at, bool bnf32, int act, bool acc,
                        const float* A, const void* B, const float* bias, float* C,
                        int M, int N, int K, int ldc) {
  dim3 grid((N + TBN - 1) / TBN, (M + TBM - 1) / TBM), blk(256);
  if (at)          k_gemm<true,  true,  0, false><<<grid, blk, 0, st>>>(A, B, bias, C, M, N, K, ldc);
  else if (bnf32)  k_gemm<false, true,  0, false><<<grid, blk, 0, st>>>(A, B, bias, C, M, N, K, ldc);
  else if (acc)    k_gemm<false, false, 0, true ><<<grid, blk, 0, st>>>(A, B, bias, C, M, N, K, ldc);
  else if (act)    k_gemm<false, false, 1, false><<<grid, blk, 0, st>>>(A, B, bias, C, M, N, K, ldc);
  else             k_gemm<false, false, 0, false><<<grid, blk, 0, st>>>(A, B, bias, C, M, N, K, ldc);
}

struct LinW { unsigned short* wt; const float* b; int din, dout; };
struct GruW { LinW w3, u3, w4, u4, w5, u5; };

extern "C" void kernel_launch(void* const* d_in, const int* in_sizes, int n_in,
                              void* d_out, int out_size, void* d_ws, size_t ws_size,
                              hipStream_t stream) {
  (void)in_sizes; (void)n_in; (void)out_size; (void)ws_size;

  const float* roi   = (const float*)d_in[0];
  const float* unf   = (const float*)d_in[1];
  const float* objd  = (const float*)d_in[2];
  const int*   rp    = (const int*)  d_in[3];
  const float* embE  = (const float*)d_in[4];
  const float* embP  = (const float*)d_in[5];
  const float* adjEE = (const float*)d_in[6];   // [3,151,151]
  const float* adjEP = (const float*)d_in[7];   // [2,151,51]
  const float* adjPE = (const float*)d_in[8];   // [2,51,151]
  const float* adjPP = (const float*)d_in[9];   // [3,51,51]

  char* wsBase = (char*)d_ws;
  size_t wsOff = 0;
  auto alloc = [&](size_t bytes) -> void* {
    wsOff = (wsOff + 255) & ~(size_t)255;
    void* p = wsBase + wsOff;
    wsOff += bytes;
    return p;
  };
  auto allocF = [&](size_t e) { return (float*)alloc(e * 4); };
  auto allocH = [&](size_t e) { return (unsigned short*)alloc(e * 2); };

  // ---- consume params in dict-insertion order; transpose weights -> bf16 [dout,din]
  int pi = 10;
  auto take_lin = [&](int din, int dout) -> LinW {
    const float* w = (const float*)d_in[pi++];
    const float* b = (const float*)d_in[pi++];
    unsigned short* wt = allocH((size_t)din * dout);
    dim3 g((din + 31) / 32, (dout + 31) / 32), blk(32, 8);
    k_transpose_bf16<<<g, blk, 0, stream>>>(w, wt, din, dout);
    return LinW{wt, b, din, dout};
  };
  LinW L_initE = take_lin(EMB, HIDDEN);
  LinW L_initP = take_lin(EMB, HIDDEN);
  LinW L_obj   = take_lin(4096, HIDDEN);
  LinW L_rel   = take_lin(4096, HIDDEN);
  LinW L_sOE0 = take_lin(HIDDEN, H2), L_sOE1 = take_lin(H2, H4);
  LinW L_sOP0 = take_lin(HIDDEN, H2), L_sOP1 = take_lin(H2, H4);
  LinW L_sIE0 = take_lin(HIDDEN, H2), L_sIE1 = take_lin(H2, H4);
  LinW L_sIP0 = take_lin(HIDDEN, H2), L_sIP1 = take_lin(H2, H4);
  LinW L_rOE0 = take_lin(1536, 1536), L_rOE1 = take_lin(1536, HIDDEN);
  LinW L_rOP0 = take_lin(1536, 1536), L_rOP1 = take_lin(1536, HIDDEN);
  LinW L_rIE0 = take_lin(768, 768),   L_rIE1 = take_lin(768, HIDDEN);
  LinW L_rIP0 = take_lin(768, 768),   L_rIP1 = take_lin(768, HIDDEN);
  auto take_gru = [&]() -> GruW {
    GruW g;
    g.w3 = take_lin(HIDDEN, HIDDEN); g.u3 = take_lin(HIDDEN, HIDDEN);
    g.w4 = take_lin(HIDDEN, HIDDEN); g.u4 = take_lin(HIDDEN, HIDDEN);
    g.w5 = take_lin(HIDDEN, HIDDEN); g.u5 = take_lin(HIDDEN, HIDDEN);
    return g;
  };
  GruW G_OE = take_gru(), G_OP = take_gru(), G_IE = take_gru(), G_IP = take_gru();
  LinW L_oIP0 = take_lin(HIDDEN, HIDDEN), L_oIP1 = take_lin(HIDDEN, HIDDEN);
  LinW L_oOP0 = take_lin(HIDDEN, HIDDEN), L_oOP1 = take_lin(HIDDEN, HIDDEN);
  LinW L_oIE0 = take_lin(HIDDEN, HIDDEN), L_oIE1 = take_lin(HIDDEN, HIDDEN);
  LinW L_oOE0 = take_lin(HIDDEN, HIDDEN), L_oOE1 = take_lin(HIDDEN, HIDDEN);

  // ---- activation buffers
  float* nodes_oe = allocF((size_t)NOC * HIDDEN);
  float* nodes_op = allocF((size_t)NRC * HIDDEN);
  float* nodes_ie = allocF((size_t)NEI * HIDDEN);
  float* nodes_ip = allocF((size_t)NPI * HIDDEN);
  float* edges_ie = allocF((size_t)NEI * NOC);
  float* edges_ip = allocF((size_t)NPI * NRC);
  float* send_oe  = allocF((size_t)NOC * H4);
  float* send_op  = allocF((size_t)NRC * H4);
  float* send_ie  = allocF((size_t)NEI * H4);
  float* send_ip  = allocF((size_t)NPI * H4);
  float* cat_oe   = allocF((size_t)NOC * 1536);
  float* cat_op   = allocF((size_t)NRC * 1536);
  float* cat_ie   = allocF((size_t)NEI * 768);
  float* cat_ip   = allocF((size_t)NPI * 768);
  float* msg_oe   = allocF((size_t)NOC * HIDDEN);
  float* msg_op   = allocF((size_t)NRC * HIDDEN);
  float* msg_ie   = allocF((size_t)NEI * HIDDEN);
  float* msg_ip   = allocF((size_t)NPI * HIDDEN);
  float* tmp1     = allocF((size_t)NPI * HIDDEN);  // MLP hidden
  float* Zb       = allocF((size_t)NPI * HIDDEN);
  float* Rb       = allocF((size_t)NPI * HIDDEN);
  float* Hb       = allocF((size_t)NPI * HIDDEN);
  float* q_op     = allocF((size_t)NRC * HIDDEN);
  float* q_oe     = allocF((size_t)NOC * HIDDEN);
  unsigned short* q_op_bf = allocH((size_t)NRC * HIDDEN);
  unsigned short* q_oe_bf = allocH((size_t)NOC * HIDDEN);

  float* out_ent  = (float*)d_out;                       // [4096,151]
  float* out_pred = (float*)d_out + (size_t)NEI * NOC;   // [16384,51]

  auto ew = [&](int n) { return dim3((unsigned)((n + 255) / 256)); };
  auto linear = [&](const float* x, int M, const LinW& L, float* out, int ldc, int act, bool acc) {
    launch_gemm(stream, false, false, act, acc, x, L.wt, L.b, out, M, L.dout, L.din, ldc);
  };
  auto mlp2 = [&](const float* x, int M, const LinW& L0, const LinW& L1,
                  float* out, int ldc_out, bool last_act) {
    linear(x, M, L0, tmp1, L0.dout, 1, false);
    linear(tmp1, M, L1, out, ldc_out, last_act ? 1 : 0, false);
  };
  auto do_gru = [&](const GruW& g, const float* msg, float* node, int M) {
    linear(msg,  M, g.w3, Zb, HIDDEN, 0, false);
    linear(node, M, g.u3, Zb, HIDDEN, 0, true);
    linear(msg,  M, g.w4, Rb, HIDDEN, 0, false);
    linear(node, M, g.u4, Rb, HIDDEN, 0, true);
    int n = M * HIDDEN;
    k_sig_mul<<<ew(n), 256, 0, stream>>>(Rb, node, n);      // Rb = sigmoid(Rb)*node
    linear(msg, M, g.w5, Hb, HIDDEN, 0, false);
    linear(Rb,  M, g.u5, Hb, HIDDEN, 0, true);
    k_gru_final<<<ew(n), 256, 0, stream>>>(Zb, Hb, node, n);
  };

  // ---- initial state
  k_softmax_rows<<<NEI, 32, 0, stream>>>(objd, edges_ie, NEI, NOC);
  k_fill<<<ew(NPI * NRC), 256, 0, stream>>>(edges_ip, 1.0f / (float)NRC, NPI * NRC);
  linear(embE, NOC, L_initE, nodes_oe, HIDDEN, 0, false);
  linear(embP, NRC, L_initP, nodes_op, HIDDEN, 0, false);
  linear(roi,  NEI, L_obj,   nodes_ie, HIDDEN, 0, false);
  linear(unf,  NPI, L_rel,   nodes_ip, HIDDEN, 0, false);

  for (int t = 0; t < 3; ++t) {
    // send MLPs
    mlp2(nodes_oe, NOC, L_sOE0, L_sOE1, send_oe, H4, true);
    mlp2(nodes_op, NRC, L_sOP0, L_sOP1, send_op, H4, true);
    mlp2(nodes_ie, NEI, L_sIE0, L_sIE1, send_ie, H4, true);
    mlp2(nodes_ip, NPI, L_sIP0, L_sIP1, send_ip, H4, true);

    // cat_oe = [r_e2e(3*256) | r_p2e(2*256) | r_i2e(256)]  -> msg_oe
    for (int k = 0; k < 3; ++k)
      launch_gemm(stream, true, true, 0, false, adjEE + (size_t)k * NOC * NOC, send_oe,
                  nullptr, cat_oe + k * H4, NOC, H4, NOC, 1536);
    for (int k = 0; k < 2; ++k)
      launch_gemm(stream, true, true, 0, false, adjPE + (size_t)k * NRC * NOC, send_op,
                  nullptr, cat_oe + (3 + k) * H4, NOC, H4, NRC, 1536);
    launch_gemm(stream, true, true, 0, false, edges_ie, send_ie, nullptr,
                cat_oe + 5 * H4, NOC, H4, NEI, 1536);
    mlp2(cat_oe, NOC, L_rOE0, L_rOE1, msg_oe, HIDDEN, true);

    // cat_op = [r_e2p(2*256) | r_p2p(3*256) | r_i2p(256)] -> msg_op
    for (int k = 0; k < 2; ++k)
      launch_gemm(stream, true, true, 0, false, adjEP + (size_t)k * NOC * NRC, send_oe,
                  nullptr, cat_op + k * H4, NRC, H4, NOC, 1536);
    for (int k = 0; k < 3; ++k)
      launch_gemm(stream, true, true, 0, false, adjPP + (size_t)k * NRC * NRC, send_op,
                  nullptr, cat_op + (2 + k) * H4, NRC, H4, NRC, 1536);
    launch_gemm(stream, true, true, 0, false, edges_ip, send_ip, nullptr,
                cat_op + 5 * H4, NRC, H4, NPI, 1536);
    mlp2(cat_op, NRC, L_rOP0, L_rOP1, msg_op, HIDDEN, true);

    // cat_ie = [b1 | segsum(subj) | segsum(obj)] -> msg_ie
    k_fill<<<ew(NEI * 768), 256, 0, stream>>>(cat_ie, 0.0f, NEI * 768);
    launch_gemm(stream, false, true, 0, false, edges_ie, send_oe, nullptr,
                cat_ie, NEI, H4, NOC, 768);
    k_scatter_pairs<<<ew(NPI * H4), 256, 0, stream>>>(send_ip, rp, cat_ie, NPI, H4, 768);
    mlp2(cat_ie, NEI, L_rIE0, L_rIE1, msg_ie, HIDDEN, true);

    // cat_ip = [c1 | send_ie[subj] | send_ie[obj]] -> msg_ip
    launch_gemm(stream, false, true, 0, false, edges_ip, send_op, nullptr,
                cat_ip, NPI, H4, NRC, 768);
    k_gather_pairs<<<ew(NPI * H4), 256, 0, stream>>>(send_ie, rp, cat_ip, NPI, H4, 768);
    mlp2(cat_ip, NPI, L_rIP0, L_rIP1, msg_ip, HIDDEN, true);

    // GRUs
    do_gru(G_OE, msg_oe, nodes_oe, NOC);
    do_gru(G_OP, msg_op, nodes_op, NRC);
    do_gru(G_IE, msg_ie, nodes_ie, NEI);
    do_gru(G_IP, msg_ip, nodes_ip, NPI);

    // pred logits = out_img_pred(nodes_ip) @ out_ont_pred(nodes_op)^T
    mlp2(nodes_ip, NPI, L_oIP0, L_oIP1, msg_ip, HIDDEN, false);   // reuse msg_ip as P
    mlp2(nodes_op, NRC, L_oOP0, L_oOP1, q_op, HIDDEN, false);
    k_convert_bf16<<<ew(NRC * HIDDEN), 256, 0, stream>>>(q_op, q_op_bf, NRC * HIDDEN);
    launch_gemm(stream, false, false, 0, false, msg_ip, q_op_bf, nullptr,
                out_pred, NPI, NRC, HIDDEN, NRC);
    k_softmax_rows<<<NPI, 32, 0, stream>>>(out_pred, edges_ip, NPI, NRC);

    // ent logits = out_img_ent(nodes_ie) @ out_ont_ent(nodes_oe)^T
    mlp2(nodes_ie, NEI, L_oIE0, L_oIE1, msg_ie, HIDDEN, false);   // reuse msg_ie as P
    mlp2(nodes_oe, NOC, L_oOE0, L_oOE1, q_oe, HIDDEN, false);
    k_convert_bf16<<<ew(NOC * HIDDEN), 256, 0, stream>>>(q_oe, q_oe_bf, NOC * HIDDEN);
    launch_gemm(stream, false, false, 0, false, msg_ie, q_oe_bf, nullptr,
                out_ent, NEI, NOC, HIDDEN, NOC);
    k_softmax_rows<<<NEI, 32, 0, stream>>>(out_ent, edges_ie, NEI, NOC);
  }
}
